// MultiHeadAttentionBlock_82214263980076
// MI455X (gfx1250) — matrix operand order repaired and checked by reading, not compiled, not run on hardware
//
#include <hip/hip_runtime.h>
#include <hip/hip_bf16.h>

// ---------------------------------------------------------------------------
// MHA block for MI455X (gfx1250, wave32, WMMA bf16 16x16x32, f32 accum)
// B=2, S=2048, D=1024, H=16, DH=64
//
// Pipeline:
//   0) f32 -> bf16 pre-convert of X (q,k,v inputs) and all weights
//   1) QKV projection GEMMs: bf16 A from global, bf16 W tile staged in LDS
//      via GLOBAL_LOAD_ASYNC_TO_LDS_B128 (ASYNCcnt), 4x WMMA per k-step
//   2) causal flash attention (online softmax, LDS transpose for P)
//   3) output projection (same staged-LDS GEMM), f32 output
// ---------------------------------------------------------------------------

#define BATCH 2
#define SEQ   2048
#define DMODEL 1024
#define NHEAD 16
#define DHEAD 64

typedef __attribute__((ext_vector_type(16))) __bf16 v16bf;
typedef __attribute__((ext_vector_type(8)))  float  v8f;
typedef __attribute__((ext_vector_type(4)))  int    v4i;

union Frag16 {
    v16bf  v;
    uint4  u[2];
    __bf16 h[16];
};

__device__ __forceinline__ uint4 ld16g(const __bf16* p) {
    return *reinterpret_cast<const uint4*>(p);
}

__device__ __forceinline__ void cvt_f4x2(__bf16* d, float4 a, float4 b) {
    d[0] = (__bf16)a.x; d[1] = (__bf16)a.y; d[2] = (__bf16)a.z; d[3] = (__bf16)a.w;
    d[4] = (__bf16)b.x; d[5] = (__bf16)b.y; d[6] = (__bf16)b.z; d[7] = (__bf16)b.w;
}

__device__ __forceinline__ v8f wmma_bf16(const v16bf& a, const v16bf& b, v8f c) {
    return __builtin_amdgcn_wmma_f32_16x16x32_bf16(
        /*neg_a=*/false, a, /*neg_b=*/false, b,
        /*c_mod=*/(short)0, c, /*reuse_a=*/false, /*reuse_b=*/false);
}

// --------------------------- async LDS staging ------------------------------
#if __has_builtin(__builtin_amdgcn_global_load_async_to_lds_b128)
#define HAS_ASYNC_LDS 1
#else
#define HAS_ASYNC_LDS 0
#endif

typedef __attribute__((address_space(1))) v4i as1_v4i;   // global int4
typedef __attribute__((address_space(3))) v4i as3_v4i;   // LDS int4

__device__ __forceinline__ void copy16_to_lds(const __bf16* g, __bf16* l) {
#if HAS_ASYNC_LDS
    as1_v4i* gp = (as1_v4i*)(g);
    as3_v4i* lp = (as3_v4i*)(l);
    __builtin_amdgcn_global_load_async_to_lds_b128(gp, lp, 0, 0);
#else
    *reinterpret_cast<uint4*>(l) = *reinterpret_cast<const uint4*>(g);
#endif
}

__device__ __forceinline__ void async_copy_wait() {
#if HAS_ASYNC_LDS
#if __has_builtin(__builtin_amdgcn_s_wait_asynccnt)
    __builtin_amdgcn_s_wait_asynccnt(0);
#else
    asm volatile("s_wait_asynccnt 0x0" ::: "memory");
#endif
#endif
}

// ---------------------------------------------------------------------------
// Kernel 0: elementwise f32 -> bf16 (8 elements / thread, b128 store)
// ---------------------------------------------------------------------------
__global__ __launch_bounds__(256)
void f32_to_bf16_kernel(const float* __restrict__ src, __bf16* __restrict__ dst,
                        int n8)
{
    const int i = blockIdx.x * 256 + threadIdx.x;
    if (i < n8) {
        const float4* s4 = reinterpret_cast<const float4*>(src) + 2 * (size_t)i;
        Frag16 f;
        cvt_f4x2(f.h, s4[0], s4[1]);
        reinterpret_cast<uint4*>(dst)[i] = f.u[0];
    }
}

// ---------------------------------------------------------------------------
// Kernel 1: QKV projections. C = X @ W^T, all-bf16 operands.
// Per block: 8 waves, one 16-row tile each, shared 64-col group.
// W tile (64 cols x 32 k, 4KB) staged in LDS by async copy each k-step.
// A-frag: lane (m=l&15, half=l>>4): h[0..7]=X[m][k+8h..], h[8..15]=X[m][k+16+8h..]
// B-frag from LDS tile row (t*16+ln): h[0..7]=W[n][k+8h..], h[8..15]=W[n][k+16+8h..]
// Outputs: Qb,Kb bf16 [B,H,S,DH]; V bf16 [B,H,DH,S] (transposed for PV).
// ---------------------------------------------------------------------------
__global__ __launch_bounds__(256)
void qkv_proj_kernel(const __bf16* __restrict__ Xq, const __bf16* __restrict__ Xk,
                     const __bf16* __restrict__ Xv,
                     const __bf16* __restrict__ Wq, const __bf16* __restrict__ Wk,
                     const __bf16* __restrict__ Wv,
                     __bf16* __restrict__ Qb, __bf16* __restrict__ Kb,
                     __bf16* __restrict__ Vt)
{
    __shared__ __align__(16) __bf16 wtile[64 * 32];   // 4KB W tile

    const int tid  = threadIdx.x;
    const int lane = tid & 31;
    const int wave = tid >> 5;
    const int half = lane >> 4;
    const int ln   = lane & 15;

    const int rowtile = blockIdx.x * 8 + wave;    // 0..255 (B*S/16 tiles)
    const int colbase = blockIdx.y * 64;          // 16 groups of 64 cols
    const int p       = blockIdx.z;               // 0=Q 1=K 2=V

    const __bf16* X = (p == 0) ? Xq : ((p == 1) ? Xk : Xv);
    const __bf16* W = (p == 0) ? Wq : ((p == 1) ? Wk : Wv);

    const __bf16* xr = X + (size_t)(rowtile * 16 + ln) * DMODEL;
    // this thread's slice of the staged W tile copy (16B each, 256*16B = 4KB)
    const int crow = tid >> 2;          // 0..63
    const int cseg = tid & 3;           // 0..3
    const __bf16* wsrc0 = W + (size_t)(colbase + crow) * DMODEL + cseg * 8;
    __bf16* wdst = wtile + crow * 32 + cseg * 8;

    v8f acc[4];
#pragma unroll
    for (int t = 0; t < 4; ++t) acc[t] = v8f{0.f, 0.f, 0.f, 0.f, 0.f, 0.f, 0.f, 0.f};

    for (int k = 0; k < DMODEL; k += 32) {
        __syncthreads();                        // previous tile fully consumed
        copy16_to_lds(wsrc0 + k, wdst);         // async 4KB stage
        __builtin_prefetch(xr + k + 32, 0, 1);  // next A chunk -> global_prefetch
        Frag16 a;
        a.u[0] = ld16g(xr + k + half * 8);
        a.u[1] = ld16g(xr + k + 16 + half * 8);
        async_copy_wait();
        __syncthreads();                        // tile visible to all waves
#pragma unroll
        for (int t = 0; t < 4; ++t) {
            const __bf16* wl = wtile + (t * 16 + ln) * 32 + half * 8;
            Frag16 bf;
            bf.u[0] = *reinterpret_cast<const uint4*>(wl);
            bf.u[1] = *reinterpret_cast<const uint4*>(wl + 16);
            acc[t] = wmma_bf16(a.v, bf.v, acc[t]);
        }
    }

    // C-frag layout: lane (half, ln), VGPR r -> (m = r + 8*half, n = ln)
#pragma unroll
    for (int t = 0; t < 4; ++t) {
#pragma unroll
        for (int r = 0; r < 8; ++r) {
            const int token = rowtile * 16 + r + 8 * half;
            const int n     = colbase + t * 16 + ln;
            const int b  = token >> 11;          // /SEQ
            const int s  = token & (SEQ - 1);
            const int hh = n >> 6;               // /DHEAD
            const int dh = n & (DHEAD - 1);
            const __bf16 val = (__bf16)acc[t][r];
            if (p == 0)
                Qb[(((size_t)(b * NHEAD + hh)) * SEQ + s) * DHEAD + dh] = val;
            else if (p == 1)
                Kb[(((size_t)(b * NHEAD + hh)) * SEQ + s) * DHEAD + dh] = val;
            else
                Vt[(((size_t)(b * NHEAD + hh)) * DHEAD + dh) * SEQ + s] = val;
        }
    }
}

// ---------------------------------------------------------------------------
// Kernel 2: causal flash attention. One wave per (b,h,q-tile of 16).
// 32 keys per step: S = Q Kchunk^T via 4 WMMAs, online softmax, P transposed
// C-layout -> A-layout through per-wave LDS, then 4 WMMAs for P @ V.
// ---------------------------------------------------------------------------
__global__ __launch_bounds__(128)
void flash_attn_kernel(const __bf16* __restrict__ Qb, const __bf16* __restrict__ Kb,
                       const __bf16* __restrict__ Vt, __bf16* __restrict__ Ob)
{
    __shared__ __align__(16) __bf16 plds[4 * 16 * 32];   // 1KB per wave

    const int lane = threadIdx.x & 31;
    const int wave = threadIdx.x >> 5;
    const int half = lane >> 4;
    const int ln   = lane & 15;

    const int bh = blockIdx.y;           // 0..31
    const int b  = bh >> 4;
    const int hh = bh & 15;
    const int qtile = blockIdx.x * 4 + wave;   // 0..127
    const int qbase = qtile * 16;

    const __bf16* Qh = Qb + (size_t)bh * SEQ * DHEAD;
    const __bf16* Kh = Kb + (size_t)bh * SEQ * DHEAD;
    const __bf16* Vh = Vt + (size_t)bh * DHEAD * SEQ;

    // Q A-fragments for d = 0..31 and 32..63
    Frag16 qf[2];
#pragma unroll
    for (int c = 0; c < 2; ++c) {
        const __bf16* qp = Qh + (size_t)(qbase + ln) * DHEAD + c * 32 + half * 16;
        qf[c].u[0] = ld16g(qp);
        qf[c].u[1] = ld16g(qp + 8);
    }

    v8f o[4];
#pragma unroll
    for (int t = 0; t < 4; ++t) o[t] = v8f{0.f, 0.f, 0.f, 0.f, 0.f, 0.f, 0.f, 0.f};
    float rmax[8], rsum[8];
#pragma unroll
    for (int r = 0; r < 8; ++r) { rmax[r] = -3.0e38f; rsum[r] = 0.f; }

    __bf16* pw = plds + wave * 512;
    const int nj32 = (qtile + 2) >> 1;   // ceil((qtile+1)/2) key-pair steps

    for (int j32 = 0; j32 < nj32; ++j32) {
        const int kb0 = j32 * 32;
        v8f sc[2];
#pragma unroll
        for (int sub = 0; sub < 2; ++sub) {
            const int kt = kb0 + sub * 16;
            // K B-frags: lane column = key token kt+ln, contiguous 16 bf16 along d
            const __bf16* kp = Kh + (size_t)(kt + ln) * DHEAD + half * 16;
            __builtin_prefetch(kp + 32 * DHEAD, 0, 1);    // next key block
            Frag16 k0, k1;
            k0.u[0] = ld16g(kp);      k0.u[1] = ld16g(kp + 8);
            k1.u[0] = ld16g(kp + 32); k1.u[1] = ld16g(kp + 40);
            v8f s = v8f{0.f, 0.f, 0.f, 0.f, 0.f, 0.f, 0.f, 0.f};
            s = wmma_bf16(qf[0].v, k0.v, s);
            s = wmma_bf16(qf[1].v, k1.v, s);
            // scale (1/sqrt(64)) + causal mask
#pragma unroll
            for (int r = 0; r < 8; ++r) {
                const int qg = qbase + r + 8 * half;
                const int kg = kt + ln;
                const float v = s[r] * 0.125f;
                s[r] = (kg <= qg) ? v : -3.0e38f;
            }
            sc[sub] = s;
        }
        // online softmax per row (rows live in 16-lane halves; xor 1/2/4/8 stays in-half)
#pragma unroll
        for (int r = 0; r < 8; ++r) {
            float mx = fmaxf(sc[0][r], sc[1][r]);
            mx = fmaxf(mx, __shfl_xor(mx, 1, 32));
            mx = fmaxf(mx, __shfl_xor(mx, 2, 32));
            mx = fmaxf(mx, __shfl_xor(mx, 4, 32));
            mx = fmaxf(mx, __shfl_xor(mx, 8, 32));
            const float nm    = fmaxf(rmax[r], mx);
            const float alpha = __expf(rmax[r] - nm);
            rmax[r] = nm;
            const float p0 = __expf(sc[0][r] - nm);
            const float p1 = __expf(sc[1][r] - nm);
            sc[0][r] = p0; sc[1][r] = p1;
            float rs = p0 + p1;
            rs += __shfl_xor(rs, 1, 32);
            rs += __shfl_xor(rs, 2, 32);
            rs += __shfl_xor(rs, 4, 32);
            rs += __shfl_xor(rs, 8, 32);
            rsum[r] = rsum[r] * alpha + rs;
            o[0][r] *= alpha; o[1][r] *= alpha; o[2][r] *= alpha; o[3][r] *= alpha;
        }
        // transpose P: C-layout -> 16x32 row-major in LDS -> A-layout
#pragma unroll
        for (int r = 0; r < 8; ++r) {
            const int m = r + 8 * half;
            pw[m * 32 + ln]      = (__bf16)sc[0][r];
            pw[m * 32 + 16 + ln] = (__bf16)sc[1][r];
        }
        asm volatile("s_wait_dscnt 0x0" ::: "memory");
        Frag16 pf;
        {
            const __bf16* pr = pw + ln * 32 + half * 16;
            pf.u[0] = *reinterpret_cast<const uint4*>(pr);
            pf.u[1] = *reinterpret_cast<const uint4*>(pr + 8);
        }
        // P @ V : V stored [B,H,DH,S] so B-frag reads are contiguous along S
#pragma unroll
        for (int t = 0; t < 4; ++t) {
            const __bf16* vp = Vh + (size_t)(t * 16 + ln) * SEQ + kb0 + half * 16;
            Frag16 vf;
            vf.u[0] = ld16g(vp);
            vf.u[1] = ld16g(vp + 8);
            o[t] = wmma_bf16(pf.v, vf.v, o[t]);
        }
    }

    // epilogue: normalize and store [B,S,D] bf16 (token-major for out proj)
#pragma unroll
    for (int t = 0; t < 4; ++t) {
#pragma unroll
        for (int r = 0; r < 8; ++r) {
            const int srow = qbase + r + 8 * half;
            const int col  = hh * 64 + t * 16 + ln;
            const float val = o[t][r] / rsum[r];
            Ob[((size_t)(b * SEQ + srow)) * DMODEL + col] = (__bf16)val;
        }
    }
}

// ---------------------------------------------------------------------------
// Kernel 3: output projection Y = Ob @ Wo^T (all-bf16 operands, f32 out),
// same LDS-staged weight tile scheme as kernel 1.
// ---------------------------------------------------------------------------
__global__ __launch_bounds__(256)
void out_proj_kernel(const __bf16* __restrict__ Ob, const __bf16* __restrict__ Wo,
                     float* __restrict__ Y)
{
    __shared__ __align__(16) __bf16 wtile[64 * 32];

    const int tid  = threadIdx.x;
    const int lane = tid & 31;
    const int wave = tid >> 5;
    const int half = lane >> 4;
    const int ln   = lane & 15;

    const int rowtile = blockIdx.x * 8 + wave;
    const int colbase = blockIdx.y * 64;

    const __bf16* orow = Ob + (size_t)(rowtile * 16 + ln) * DMODEL;
    const int crow = tid >> 2;
    const int cseg = tid & 3;
    const __bf16* wsrc0 = Wo + (size_t)(colbase + crow) * DMODEL + cseg * 8;
    __bf16* wdst = wtile + crow * 32 + cseg * 8;

    v8f acc[4];
#pragma unroll
    for (int t = 0; t < 4; ++t) acc[t] = v8f{0.f, 0.f, 0.f, 0.f, 0.f, 0.f, 0.f, 0.f};

    for (int k = 0; k < DMODEL; k += 32) {
        __syncthreads();
        copy16_to_lds(wsrc0 + k, wdst);
        __builtin_prefetch(orow + k + 32, 0, 1);
        Frag16 a;
        a.u[0] = ld16g(orow + k + half * 8);
        a.u[1] = ld16g(orow + k + 16 + half * 8);
        async_copy_wait();
        __syncthreads();
#pragma unroll
        for (int t = 0; t < 4; ++t) {
            const __bf16* wl = wtile + (t * 16 + ln) * 32 + half * 8;
            Frag16 bf;
            bf.u[0] = *reinterpret_cast<const uint4*>(wl);
            bf.u[1] = *reinterpret_cast<const uint4*>(wl + 16);
            acc[t] = wmma_bf16(a.v, bf.v, acc[t]);
        }
    }

#pragma unroll
    for (int t = 0; t < 4; ++t) {
#pragma unroll
        for (int r = 0; r < 8; ++r) {
            const int token = rowtile * 16 + r + 8 * half;
            const int n     = colbase + t * 16 + ln;
            Y[(size_t)token * DMODEL + n] = acc[t][r];
        }
    }
}

// ---------------------------------------------------------------------------
// Host launcher
// ---------------------------------------------------------------------------
extern "C" void kernel_launch(void* const* d_in, const int* in_sizes, int n_in,
                              void* d_out, int out_size, void* d_ws, size_t ws_size,
                              hipStream_t stream)
{
    const float* Xq = (const float*)d_in[0];
    const float* Xk = (const float*)d_in[1];
    const float* Xv = (const float*)d_in[2];
    // d_in[3] = causal mask (int32) -- causality computed analytically
    const float* Wq = (const float*)d_in[4];
    const float* Wk = (const float*)d_in[5];
    const float* Wv = (const float*)d_in[6];
    const float* Wo = (const float*)d_in[7];

    char* ws = (char*)d_ws;
    const size_t MB = (size_t)1 << 20;
    __bf16* Qb  = (__bf16*)(ws);             // [B,H,S,DH]  8MB
    __bf16* Kb  = (__bf16*)(ws + 8  * MB);   // [B,H,S,DH]  8MB
    __bf16* Vt  = (__bf16*)(ws + 16 * MB);   // [B,H,DH,S]  8MB
    __bf16* Ob  = (__bf16*)(ws + 24 * MB);   // [B,S,D]     8MB
    __bf16* Xqb = (__bf16*)(ws + 32 * MB);   // [B*S,D]     8MB
    __bf16* Xkb = (__bf16*)(ws + 40 * MB);
    __bf16* Xvb = (__bf16*)(ws + 48 * MB);
    __bf16* Wqb = (__bf16*)(ws + 56 * MB);   // [D,D]       2MB each
    __bf16* Wkb = (__bf16*)(ws + 58 * MB);
    __bf16* Wvb = (__bf16*)(ws + 60 * MB);
    __bf16* Wob = (__bf16*)(ws + 62 * MB);
    float* Y = (float*)d_out;

    // 0) pre-convert everything to bf16 (one-time, bandwidth-trivial)
    const int n8x = (BATCH * SEQ * DMODEL) / 8;   // 524288
    const int n8w = (DMODEL * DMODEL) / 8;        // 131072
    f32_to_bf16_kernel<<<n8x / 256, 256, 0, stream>>>(Xq, Xqb, n8x);
    f32_to_bf16_kernel<<<n8x / 256, 256, 0, stream>>>(Xk, Xkb, n8x);
    f32_to_bf16_kernel<<<n8x / 256, 256, 0, stream>>>(Xv, Xvb, n8x);
    f32_to_bf16_kernel<<<n8w / 256, 256, 0, stream>>>(Wq, Wqb, n8w);
    f32_to_bf16_kernel<<<n8w / 256, 256, 0, stream>>>(Wk, Wkb, n8w);
    f32_to_bf16_kernel<<<n8w / 256, 256, 0, stream>>>(Wv, Wvb, n8w);
    f32_to_bf16_kernel<<<n8w / 256, 256, 0, stream>>>(Wo, Wob, n8w);

    // 1) QKV projections: 256 row tiles x 16 col groups x {Q,K,V}
    qkv_proj_kernel<<<dim3(32, 16, 3), 256, 0, stream>>>(Xqb, Xkb, Xvb,
                                                         Wqb, Wkb, Wvb,
                                                         Qb, Kb, Vt);
    // 2) causal flash attention: 128 q-tiles (4 waves/block) x 32 (b,h)
    flash_attn_kernel<<<dim3(32, 32), 128, 0, stream>>>(Qb, Kb, Vt, Ob);
    // 3) output projection
    out_proj_kernel<<<dim3(32, 16), 256, 0, stream>>>(Ob, Wob, Y);
}
